// TreeTransformerCell_20607253086593
// MI455X (gfx1250) — matrix operand order repaired and checked by reading, not compiled, not run on hardware
//
#include <hip/hip_runtime.h>
#include <hip/hip_bf16.h>

// ---------------------------------------------------------------------------
// Tree-Transformer cell, fused, bf16 WMMA (gfx1250 / CDNA5, wave32)
// ---------------------------------------------------------------------------

typedef __attribute__((ext_vector_type(16))) __bf16 v16bf;
typedef __attribute__((ext_vector_type(8)))  __bf16 v8bf;
typedef __attribute__((ext_vector_type(8)))  float  v8f;

#define DIM      256
#define NHEAD    8
#define DKH      32
#define NCHILD   8
#define FFD      1024
#define TILE_N   16
#define NTHREADS 256

// Async global->LDS path (gfx1250). Guarded so the file still compiles if the
// toolchain doesn't declare the builtin (falls back to load+cvt+ds_store).
#if defined(__AMDGCN__) && defined(__has_builtin)
#if __has_builtin(__builtin_amdgcn_global_load_async_to_lds_b128)
#define USE_ASYNC_GATHER 1
#endif
#if __has_builtin(__builtin_amdgcn_s_wait_asynccnt)
#define HAVE_WAIT_ASYNC_BUILTIN 1
#endif
#endif
#ifndef USE_ASYNC_GATHER
#define USE_ASYNC_GATHER 0
#endif
#ifndef HAVE_WAIT_ASYNC_BUILTIN
#define HAVE_WAIT_ASYNC_BUILTIN 0
#endif

#if USE_ASYNC_GATHER
// Parameter types per hipcc diagnostic: <4 x i32> pointers with explicit
// address spaces (1 = global source, 3 = LDS destination).
typedef int v4i_vs __attribute__((vector_size(4 * sizeof(int))));
typedef __attribute__((address_space(1))) v4i_vs as1_v4i;
typedef __attribute__((address_space(3))) v4i_vs as3_v4i;
#endif

// workspace layout (bf16 elements)
#define OFF_WQ 0
#define OFF_WK (DIM*DIM)
#define OFF_WV (2*DIM*DIM)
#define OFF_W1 (3*DIM*DIM)
#define OFF_W2 (3*DIM*DIM + FFD*DIM)

// dynamic LDS layout (byte offsets, all 16B aligned)
#define SMO_X    0        // 128 x 256 bf16  = 65536
#define SMO_K    65536    // 128 x 256 bf16  = 65536  (phase1: f32 gather staging)
#define SMO_V    131072   // 128 x 256 bf16  = 65536  (phase1: f32 gather staging)
#define SMO_P    196608   // 16  x 256 bf16  = 8192
#define SMO_QF   204800   // 16  x 256 f32   = 16384
#define SMO_CTXF 221184   // 16  x 256 f32   = 16384 (attn+res; reused for FFN out)
#define SMO_CNF  237568   // 16  x 256 f32   = 16384 (post-LN1, residual source)
#define SMO_CNB  253952   // 16  x 256 bf16  = 8192
#define SMO_HID  262144   // 16  x 1024 bf16 = 32768
#define SMO_IDS  294912   // 128 int         = 512
#define SMO_RED  295424   // 16 x 16 f32     = 1024
#define SMO_MV   296448   // 16 f32          = 64
#define SMEM_BYTES 296512

// ---- WMMA fragment helpers (layouts per CDNA5 ISA 7.12.2) ------------------

// A: 16-bit 16x32.  lane<16: halves0-7 -> K+0..7, halves8-15 -> K+16..23
//                   lane>=16: halves0-7 -> K+8..15, halves8-15 -> K+24..31
__device__ __forceinline__ v16bf frag_a(const __bf16* base, int ld,
                                        int mbase, int kbase, int lane) {
  const int hi = lane >> 4;
  const __bf16* p = base + (mbase + (lane & 15)) * ld + kbase + hi * 8;
  v8bf lo = *(const v8bf*)p;
  v8bf up = *(const v8bf*)(p + 16);
  v16bf r;
#pragma unroll
  for (int i = 0; i < 8; ++i) { r[i] = lo[i]; r[i + 8] = up[i]; }
  return r;
}

// B: 32x16, column n = (lane&15); lanes 0-15 hold K 0..7,16..23, lanes 16-31 K 8..15,24..31.
// Weight W is [out=n][in=k] row-major, so a column of B is a contiguous W row.
__device__ __forceinline__ v16bf frag_b(const __bf16* base, int ld,
                                        int nbase, int kbase, int lane) {
  const __bf16* p = base + (size_t)(nbase + (lane & 15)) * ld
                         + kbase + (lane >> 4) * 8;
  v8bf a = *(const v8bf*)p;
  v8bf b = *(const v8bf*)(p + 16);
  v16bf r;
#pragma unroll
  for (int i = 0; i < 8; ++i) { r[i] = a[i]; r[i + 8] = b[i]; }
  return r;
}

__device__ __forceinline__ v8f wmma_bf16(v16bf a, v16bf b, v8f c) {
  return __builtin_amdgcn_wmma_f32_16x16x32_bf16(
      /*neg_a=*/false, a, /*neg_b=*/false, b,
      /*c_mod=*/(short)0, c, /*reuse_a=*/false, /*reuse_b=*/false);
}

// LayerNorm over 16 rows x 256 cols living in LDS (16 threads per row).
__device__ __forceinline__ void layernorm16(const float* src, float* red,
                                            float* mv, const float* g,
                                            const float* b, float* dstF,
                                            __bf16* dstB, float* gout,
                                            int n0, int tid) {
  const int n = tid >> 4;
  const int i = tid & 15;
  float s = 0.f;
#pragma unroll
  for (int j = 0; j < 16; ++j) s += src[n * DIM + i + j * 16];
  red[n * 16 + i] = s;
  __syncthreads();
  if (tid < 16) {
    float m = 0.f;
#pragma unroll
    for (int k = 0; k < 16; ++k) m += red[tid * 16 + k];
    mv[tid] = m * (1.0f / DIM);
  }
  __syncthreads();
  const float mu = mv[n];
  float s2 = 0.f;
#pragma unroll
  for (int j = 0; j < 16; ++j) {
    float d = src[n * DIM + i + j * 16] - mu;
    s2 += d * d;
  }
  red[n * 16 + i] = s2;
  __syncthreads();
  if (tid < 16) {
    float v = 0.f;
#pragma unroll
    for (int k = 0; k < 16; ++k) v += red[tid * 16 + k];
    mv[tid] = rsqrtf(v * (1.0f / DIM) + 1e-5f);
  }
  __syncthreads();
  const float rstd = mv[n];
#pragma unroll
  for (int j = 0; j < 16; ++j) {
    int d = i + j * 16;
    float y = (src[n * DIM + d] - mu) * rstd * g[d] + b[d];
    if (dstF) dstF[n * DIM + d] = y;
    if (dstB) dstB[n * DIM + d] = (__bf16)y;
    if (gout) gout[(size_t)(n0 + n) * DIM + d] = y;
  }
  __syncthreads();
}

// ---- weight fp32 -> bf16 pre-pass ------------------------------------------
__global__ void cvt_bf16_kernel(const float* __restrict__ src,
                                __bf16* __restrict__ dst, int n) {
  int i = blockIdx.x * blockDim.x + threadIdx.x;
  if (i < n) dst[i] = (__bf16)src[i];
}

// ---- main fused kernel ------------------------------------------------------
__global__ void __launch_bounds__(NTHREADS, 1)
tree_cell_kernel(const float* __restrict__ h_all,
                 const float* __restrict__ parent_h,
                 const int* __restrict__ child_ids,
                 const __bf16* __restrict__ wsb,
                 const float* __restrict__ bk, const float* __restrict__ bq,
                 const float* __restrict__ bv, const float* __restrict__ b1,
                 const float* __restrict__ b2, const float* __restrict__ ln_g,
                 const float* __restrict__ ln_b, float* __restrict__ out) {
  extern __shared__ __align__(16) char smem[];
  __bf16* Xl   = (__bf16*)(smem + SMO_X);
  __bf16* Kl   = (__bf16*)(smem + SMO_K);
  __bf16* Vl   = (__bf16*)(smem + SMO_V);
  __bf16* Pl   = (__bf16*)(smem + SMO_P);
  float*  Qf   = (float*)(smem + SMO_QF);
  float*  CTXf = (float*)(smem + SMO_CTXF);
  float*  CNf  = (float*)(smem + SMO_CNF);
  __bf16* CNb  = (__bf16*)(smem + SMO_CNB);
  __bf16* HIDb = (__bf16*)(smem + SMO_HID);
  int*    ids  = (int*)(smem + SMO_IDS);
  float*  red  = (float*)(smem + SMO_RED);
  float*  mv   = (float*)(smem + SMO_MV);

  const int tid  = threadIdx.x;
  const int lane = tid & 31;
  const int wave = tid >> 5;
  const int n0   = blockIdx.x * TILE_N;

  const __bf16* Wq = wsb + OFF_WQ;
  const __bf16* Wk = wsb + OFF_WK;
  const __bf16* Wv = wsb + OFF_WV;
  const __bf16* W1 = wsb + OFF_W1;
  const __bf16* W2 = wsb + OFF_W2;

  // ---- phase 1: gather children + parent tile into LDS (f32 -> bf16) ----
  if (tid < TILE_N * NCHILD) ids[tid] = child_ids[n0 * NCHILD + tid];
  __syncthreads();
#pragma unroll 2
  for (int idx = tid; idx < TILE_N * (DIM / 4); idx += NTHREADS) {
    int r = idx >> 6, c4 = idx & 63;
    float4 v = ((const float4*)(parent_h + (size_t)(n0 + r) * DIM))[c4];
    __bf16* d = Pl + r * DIM + c4 * 4;
    d[0] = (__bf16)v.x; d[1] = (__bf16)v.y;
    d[2] = (__bf16)v.z; d[3] = (__bf16)v.w;
  }

#if USE_ASYNC_GATHER
  {
    // Async DMA of the 128 gathered f32 rows straight into LDS (no VGPR
    // round-trip); staging buffer overlays the not-yet-live K/V region.
    float* Xstage = (float*)(smem + SMO_K);  // 128 x 256 f32 = 128KB
#pragma unroll 4
    for (int idx = tid; idx < TILE_N * NCHILD * (DIM / 4); idx += NTHREADS) {
      int r = idx >> 6, c4 = idx & 63;
      const float* src = h_all + (size_t)ids[r] * DIM + c4 * 4;
      float* dst = Xstage + r * DIM + c4 * 4;
      __builtin_amdgcn_global_load_async_to_lds_b128(
          (as1_v4i*)(const void*)src, (as3_v4i*)(void*)dst, 0, 0);
    }
#if HAVE_WAIT_ASYNC_BUILTIN
    __builtin_amdgcn_s_wait_asynccnt(0);
#else
    asm volatile("s_wait_asynccnt 0x0" ::: "memory");
#endif
    __syncthreads();
    // Convert staged f32 -> bf16 X tile (LDS -> LDS).
#pragma unroll 4
    for (int j = tid; j < TILE_N * NCHILD * (DIM / 4); j += NTHREADS) {
      float4 v = ((const float4*)Xstage)[j];
      __bf16* d = Xl + j * 4;
      d[0] = (__bf16)v.x; d[1] = (__bf16)v.y;
      d[2] = (__bf16)v.z; d[3] = (__bf16)v.w;
    }
  }
#else
#pragma unroll 4
  for (int idx = tid; idx < TILE_N * NCHILD * (DIM / 4); idx += NTHREADS) {
    int r = idx >> 6, c4 = idx & 63;
    float4 v = ((const float4*)(h_all + (size_t)ids[r] * DIM))[c4];
    __bf16* d = Xl + r * DIM + c4 * 4;
    d[0] = (__bf16)v.x; d[1] = (__bf16)v.y;
    d[2] = (__bf16)v.z; d[3] = (__bf16)v.w;
  }
#endif
  __syncthreads();

  // ---- phase 2: Q = parent @ Wq^T + bq   (M=16, N=256, K=256) ----
  // Two N-tiles per wave share every A fragment.
  {
    const int nt0 = wave * 2, nt1 = wave * 2 + 1;
    v8f acc0 = {}, acc1 = {};
#pragma unroll
    for (int kt = 0; kt < 8; ++kt) {
      v16bf a  = frag_a(Pl, DIM, 0, kt * 32, lane);
      v16bf b0 = frag_b(Wq, DIM, nt0 * 16, kt * 32, lane);
      v16bf b1 = frag_b(Wq, DIM, nt1 * 16, kt * 32, lane);
      acc0 = wmma_bf16(a, b0, acc0);
      acc1 = wmma_bf16(a, b1, acc1);
    }
    const int hi = lane >> 4;
    const int na = nt0 * 16 + (lane & 15), nb = nt1 * 16 + (lane & 15);
    const float ba = bq[na], bb2 = bq[nb];
#pragma unroll
    for (int i = 0; i < 8; ++i) {
      Qf[(i + 8 * hi) * DIM + na] = acc0[i] + ba;
      Qf[(i + 8 * hi) * DIM + nb] = acc1[i] + bb2;
    }
  }
  __syncthreads();

  // ---- phase 3: K,V = X @ W^T + b   (M=128, N=256, K=256, x2) ----
  // Each task covers an N-tile *pair* so the A fragment is loaded once.
  for (int t = wave; t < 128; t += 8) {
    const int which = t >> 6;          // 0=K, 1=V
    const int tt = t & 63;
    const int mt = tt >> 3, ntp = tt & 7;
    const int nt0 = ntp * 2, nt1 = ntp * 2 + 1;
    const __bf16* W = which ? Wv : Wk;
    const float* bb = which ? bv : bk;
    __bf16* dst = which ? Vl : Kl;
    v8f acc0 = {}, acc1 = {};
#pragma unroll
    for (int kt = 0; kt < 8; ++kt) {
      v16bf a  = frag_a(Xl, DIM, mt * 16, kt * 32, lane);
      v16bf b0 = frag_b(W, DIM, nt0 * 16, kt * 32, lane);
      v16bf b1 = frag_b(W, DIM, nt1 * 16, kt * 32, lane);
      acc0 = wmma_bf16(a, b0, acc0);
      acc1 = wmma_bf16(a, b1, acc1);
    }
    const int hi = lane >> 4;
    const int na = nt0 * 16 + (lane & 15), nb = nt1 * 16 + (lane & 15);
    const float ba = bb[na], bbv = bb[nb];
#pragma unroll
    for (int i = 0; i < 8; ++i) {
      const int m = mt * 16 + i + 8 * hi;
      dst[m * DIM + na] = (__bf16)(acc0[i] + ba);
      dst[m * DIM + nb] = (__bf16)(acc1[i] + bbv);
    }
  }
  __syncthreads();

  // ---- phase 4: per-head single-query attention + residual ----
  if (tid < TILE_N * NHEAD) {
    const int n = tid >> 3, h = tid & 7;
    const float* qv = Qf + n * DIM + h * DKH;
    float sc[NCHILD];
    float mx = -1e30f;
#pragma unroll
    for (int c = 0; c < NCHILD; ++c) {
      const __bf16* kr = Kl + (n * NCHILD + c) * DIM + h * DKH;
      float s = 0.f;
#pragma unroll
      for (int d = 0; d < DKH; ++d) s += qv[d] * (float)kr[d];
      s *= 0.17677669529663687f;  // 1/sqrt(32)
      sc[c] = s;
      mx = fmaxf(mx, s);
    }
    float den = 0.f;
#pragma unroll
    for (int c = 0; c < NCHILD; ++c) { sc[c] = expf(sc[c] - mx); den += sc[c]; }
    const float inv = 1.f / den;
#pragma unroll
    for (int d = 0; d < DKH; ++d) {
      float a = 0.f;
#pragma unroll
      for (int c = 0; c < NCHILD; ++c)
        a += sc[c] * (float)Vl[(n * NCHILD + c) * DIM + h * DKH + d];
      CTXf[n * DIM + h * DKH + d] = a * inv + qv[d];
    }
  }
  __syncthreads();

  // ---- phase 5: LayerNorm1 -> CNf (f32 residual) + CNb (bf16 GEMM A) ----
  layernorm16(CTXf, red, mv, ln_g, ln_b, CNf, CNb, nullptr, 0, tid);

  // ---- phase 6: hid = gelu(ctxn @ w1^T + b1)   (M=16, N=1024, K=256) ----
  // The A tile (16x256) is identical for all 8 N-tiles of this wave: hoist
  // all 8 A fragments into registers once.
  {
    v16bf af[8];
#pragma unroll
    for (int kt = 0; kt < 8; ++kt) af[kt] = frag_a(CNb, DIM, 0, kt * 32, lane);
    for (int nt = wave; nt < FFD / 16; nt += 8) {
      __builtin_prefetch(W1 + (size_t)((nt + 8) * 16) * DIM, 0, 1);
      v8f acc = {};
#pragma unroll
      for (int kt = 0; kt < 8; ++kt) {
        v16bf b = frag_b(W1, DIM, nt * 16, kt * 32, lane);
        acc = wmma_bf16(af[kt], b, acc);
      }
      const int n = nt * 16 + (lane & 15);
      const int hi = lane >> 4;
      const float bias = b1[n];
#pragma unroll
      for (int i = 0; i < 8; ++i) {
        float g = acc[i] + bias;
        g = 0.5f * g * (1.f + erff(g * 0.7071067811865476f));  // exact gelu
        HIDb[(i + 8 * hi) * FFD + n] = (__bf16)g;
      }
    }
  }
  __syncthreads();

  // ---- phase 7: out2 = hid @ w2^T + b2 + ctxn  (M=16, N=256, K=1024) ----
  {
    const int nt0 = wave * 2, nt1 = wave * 2 + 1;
    v8f acc0 = {}, acc1 = {};
#pragma unroll
    for (int kt = 0; kt < 32; ++kt) {
      v16bf a  = frag_a(HIDb, FFD, 0, kt * 32, lane);
      v16bf b0 = frag_b(W2, FFD, nt0 * 16, kt * 32, lane);
      v16bf b1 = frag_b(W2, FFD, nt1 * 16, kt * 32, lane);
      acc0 = wmma_bf16(a, b0, acc0);
      acc1 = wmma_bf16(a, b1, acc1);
    }
    const int hi = lane >> 4;
    const int na = nt0 * 16 + (lane & 15), nb = nt1 * 16 + (lane & 15);
    const float ba = b2[na], bbv = b2[nb];
#pragma unroll
    for (int i = 0; i < 8; ++i) {
      const int m = i + 8 * hi;
      CTXf[m * DIM + na] = acc0[i] + ba + CNf[m * DIM + na];  // reuse CTXf
      CTXf[m * DIM + nb] = acc1[i] + bbv + CNf[m * DIM + nb];
    }
  }
  __syncthreads();

  // ---- phase 8: LayerNorm2 -> global out ----
  layernorm16(CTXf, red, mv, ln_g, ln_b, nullptr, nullptr, out, n0, tid);
}

// ---------------------------------------------------------------------------
extern "C" void kernel_launch(void* const* d_in, const int* in_sizes, int n_in,
                              void* d_out, int out_size, void* d_ws,
                              size_t ws_size, hipStream_t stream) {
  const float* h_all    = (const float*)d_in[0];
  const float* parent_h = (const float*)d_in[1];
  const int*   child_id = (const int*)d_in[2];
  const float* Wk = (const float*)d_in[3];
  const float* bk = (const float*)d_in[4];
  const float* Wq = (const float*)d_in[5];
  const float* bq = (const float*)d_in[6];
  const float* Wv = (const float*)d_in[7];
  const float* bv = (const float*)d_in[8];
  const float* w1 = (const float*)d_in[9];
  const float* b1 = (const float*)d_in[10];
  const float* w2 = (const float*)d_in[11];
  const float* b2 = (const float*)d_in[12];
  const float* lg = (const float*)d_in[13];
  const float* lb = (const float*)d_in[14];
  float* out = (float*)d_out;
  __bf16* wsb = (__bf16*)d_ws;

  const int Ntot = in_sizes[1] / DIM;  // N = 50000

  // weight fp32 -> bf16 pre-pass (L2-resident afterwards)
  cvt_bf16_kernel<<<(DIM * DIM + 255) / 256, 256, 0, stream>>>(Wq, wsb + OFF_WQ, DIM * DIM);
  cvt_bf16_kernel<<<(DIM * DIM + 255) / 256, 256, 0, stream>>>(Wk, wsb + OFF_WK, DIM * DIM);
  cvt_bf16_kernel<<<(DIM * DIM + 255) / 256, 256, 0, stream>>>(Wv, wsb + OFF_WV, DIM * DIM);
  cvt_bf16_kernel<<<(FFD * DIM + 255) / 256, 256, 0, stream>>>(w1, wsb + OFF_W1, FFD * DIM);
  cvt_bf16_kernel<<<(FFD * DIM + 255) / 256, 256, 0, stream>>>(w2, wsb + OFF_W2, FFD * DIM);

  (void)hipFuncSetAttribute((const void*)tree_cell_kernel,
                            hipFuncAttributeMaxDynamicSharedMemorySize,
                            (int)SMEM_BYTES);

  dim3 grid((Ntot + TILE_N - 1) / TILE_N), block(NTHREADS);
  tree_cell_kernel<<<grid, block, SMEM_BYTES, stream>>>(
      h_all, parent_h, child_id, wsb, bk, bq, bv, b1, b2, lg, lb, out);
}
